// Qwen2MoeSparseMoeBlock_9234179686725
// MI455X (gfx1250) — matrix operand order repaired and checked by reading, not compiled, not run on hardware
//
#include <hip/hip_runtime.h>
#include <hip/hip_bf16.h>
#include <stdint.h>

typedef __bf16 bf16;
typedef __attribute__((ext_vector_type(16))) __bf16 v16bf;
typedef __attribute__((ext_vector_type(8)))  __bf16 v8bf;
typedef __attribute__((ext_vector_type(4)))  __bf16 v4bf;
typedef __attribute__((ext_vector_type(8)))  float   v8f;

namespace {
constexpr int T_TOK = 2048;   // tokens (B*S)
constexpr int D_MOD = 2048;   // hidden dim
constexpr int N_EXP = 8;      // experts
constexpr int F_EXP = 1408;   // expert intermediate
constexpr int F_SH  = 5632;   // shared intermediate
constexpr int TOPK  = 4;

constexpr int BM = 128;       // token tile
constexpr int BN = 64;        // output-col tile
constexpr int BK = 32;        // K tile == WMMA K
constexpr int LP = BK + 8;    // LDS row pitch (bf16): 80 B, keeps 16B alignment
constexpr int ASZ = BM * LP;  // one A stage (bf16 elems)
constexpr int BSZ = BN * LP;  // one B stage (bf16 elems)
}

__device__ __forceinline__ float fsigmoid(float v) { return 1.0f / (1.0f + __expf(-v)); }

// ---------------------------------------------------------------------------
// Router: logits, softmax, top-4 dense weights, shared-expert sigmoid gate.
// One wave (32 lanes) per token; 8 waves per block.
// ---------------------------------------------------------------------------
__global__ __launch_bounds__(256)
void moe_router_kernel(const float* __restrict__ x,
                       const float* __restrict__ gate_w,        // [E, D]
                       const float* __restrict__ shared_gate_w, // [D]
                       float* __restrict__ w_out,               // [T, E]
                       float* __restrict__ sgate_out,           // [T]
                       float* __restrict__ logits_out)          // [T, E]
{
  const int wave = threadIdx.x >> 5;
  const int lane = threadIdx.x & 31;
  const int t = blockIdx.x * 8 + wave;
  const float* xr = x + (size_t)t * D_MOD;

  float acc[N_EXP];
#pragma unroll
  for (int e = 0; e < N_EXP; ++e) acc[e] = 0.0f;
  float accs = 0.0f;

  for (int d = lane; d < D_MOD; d += 32) {
    const float xv = xr[d];
#pragma unroll
    for (int e = 0; e < N_EXP; ++e) acc[e] = fmaf(xv, gate_w[e * D_MOD + d], acc[e]);
    accs = fmaf(xv, shared_gate_w[d], accs);
  }
#pragma unroll
  for (int off = 16; off > 0; off >>= 1) {
#pragma unroll
    for (int e = 0; e < N_EXP; ++e) acc[e] += __shfl_xor(acc[e], off, 32);
    accs += __shfl_xor(accs, off, 32);
  }

  if (lane == 0) {
    float mx = acc[0];
#pragma unroll
    for (int e = 1; e < N_EXP; ++e) mx = fmaxf(mx, acc[e]);
    float p[N_EXP];
    float s = 0.0f;
#pragma unroll
    for (int e = 0; e < N_EXP; ++e) { p[e] = __expf(acc[e] - mx); s += p[e]; }
    const float inv = 1.0f / s;
#pragma unroll
    for (int e = 0; e < N_EXP; ++e) p[e] *= inv;

    unsigned sel = 0;
#pragma unroll
    for (int k = 0; k < TOPK; ++k) {
      int best = 0; float bv = -1.0f;
#pragma unroll
      for (int e = 0; e < N_EXP; ++e)
        if (!(sel & (1u << e)) && p[e] > bv) { bv = p[e]; best = e; }
      sel |= (1u << best);
    }
#pragma unroll
    for (int e = 0; e < N_EXP; ++e) {
      w_out[(size_t)t * N_EXP + e]      = (sel & (1u << e)) ? p[e] : 0.0f;
      logits_out[(size_t)t * N_EXP + e] = acc[e];
    }
    sgate_out[t] = fsigmoid(accs);
  }
}

// ---------------------------------------------------------------------------
// Double-buffered tiled WMMA GEMM. 256 threads = 8 waves (4M x 2N), each wave
// owns a 32x32 output patch = 2x2 fragments of 16x16 per accumulator set.
//   MODE 0: C_bf16 = silu(A@B0) * (A@B1) * rowscale   (gate/up, dual-B)
//   MODE 1: C_f32  = sum_batch A_b @ B_b              (expert down-proj)
//   MODE 2: C_f32  = addin + rowscale[m] * (A @ B0)   (shared down + combine)
// A row-major [M,K]: f32 (float4 loads + packed v4bf LDS stores) or bf16
// (async copy straight to LDS via GLOBAL_LOAD_ASYNC_TO_LDS_B128, ASYNCcnt).
// B row-major [K,N] f32: float4 loads along N, transposed bf16 stores to LDS.
// Ping-pong LDS: one barrier per k-tile; next tile's global traffic is issued
// before compute on the current tile and committed after.
// ---------------------------------------------------------------------------
template <int MODE, bool ABF16>
__global__ __launch_bounds__(256)
void wmma_gemm_kernel(const void* __restrict__ Ap, int lda,
                      const float* __restrict__ B0g, const float* __restrict__ B1g, int ldb,
                      void* __restrict__ Cp, int ldc,
                      const float* __restrict__ rowscale, int rs_stride, int rs_zstep,
                      const float* __restrict__ addin,
                      int K, int nbatch, long Abatch, long Bbatch,
                      long Bzoff, long Czoff)
{
  __shared__ alignas(16) bf16 As[2 * ASZ];
  __shared__ alignas(16) bf16 Bs0[2 * BSZ];
  __shared__ alignas(16) bf16 Bs1[(MODE == 0) ? (2 * BSZ) : 1];

  const int tid  = threadIdx.x;
  const int lane = tid & 31;
  const int wave = tid >> 5;
  const int wm = wave & 3;   // 4 waves along M
  const int wn = wave >> 2;  // 2 waves along N
  const int m0 = blockIdx.y * BM;
  const int n0 = blockIdx.x * BN;
  const int zb = blockIdx.z; // expert index for MODE 0 expert launches

  const float* B0 = B0g + (size_t)zb * Bzoff;
  const float* B1 = (MODE == 0) ? (B1g + (size_t)zb * Bzoff) : nullptr;

  v8f vzero = {};
  v8f accg[2][2];
  v8f accu[2][2];
#pragma unroll
  for (int a = 0; a < 2; ++a)
#pragma unroll
    for (int b = 0; b < 2; ++b) { accg[a][b] = vzero; accu[a][b] = vzero; }

  const int ktiles_pb = K / BK;
  const int ntiles = nbatch * ktiles_pb;

  float4 aReg[4];   // A: 4x float4 per thread (f32 path)
  float4 bReg0[2];  // B: 2x float4 per thread per matrix
  float4 bReg1[2];

  // ---- stage-issue: start global traffic for k-tile `kt` into LDS slot `sl`
  auto stage_issue = [&](int kt, int sl) {
    const int b  = kt / ktiles_pb;
    const int k0 = (kt - b * ktiles_pb) * BK;
    const char* Abase = (const char*)Ap + (size_t)b * Abatch * (ABF16 ? 2 : 4);
    const float* Bb0 = B0 + (size_t)b * Bbatch;
    if (ABF16) {
      // raw bf16 tile: async byte-copy straight into LDS (ASYNCcnt)
      const bf16* Ab = (const bf16*)Abase;
      const uint32_t ldsA = (uint32_t)(size_t)(&As[sl * ASZ]);
#pragma unroll
      for (int q = 0; q < (BM * BK) / (8 * 256); ++q) {   // 2 x 16B chunks/thread
        const int chunk = tid + 256 * q;                  // 16B = 8 bf16 per chunk
        const int r  = chunk >> 2;                        // 4 chunks per 64B row
        const int qq = chunk & 3;
        const uint64_t g = (uint64_t)(uintptr_t)(Ab + (size_t)(m0 + r) * lda + k0 + qq * 8);
        const uint32_t l = ldsA + (uint32_t)(r * LP + qq * 8) * 2u;
        asm volatile("global_load_async_to_lds_b128 %0, %1, off"
                     :: "v"(l), "v"(g) : "memory");
      }
    } else {
      const float* Af = (const float*)Abase;
#pragma unroll
      for (int p = 0; p < 4; ++p) {                        // 4 quads/thread
        const int qi = tid + 256 * p;
        const int r  = qi >> 3;                            // 8 quads per 32-elem row
        const int cq = (qi & 7) * 4;
        aReg[p] = *(const float4*)&Af[(size_t)(m0 + r) * lda + k0 + cq];
      }
    }
#pragma unroll
    for (int p = 0; p < 2; ++p) {                          // 2 quads/thread
      const int qi  = tid + 256 * p;
      const int kr  = qi >> 4;                             // 16 quads per 64-elem row
      const int ncq = (qi & 15) * 4;
      bReg0[p] = *(const float4*)&Bb0[(size_t)(k0 + kr) * ldb + n0 + ncq];
    }
    if (MODE == 0) {
      const float* Bb1 = B1 + (size_t)b * Bbatch;
#pragma unroll
      for (int p = 0; p < 2; ++p) {
        const int qi  = tid + 256 * p;
        const int kr  = qi >> 4;
        const int ncq = (qi & 15) * 4;
        bReg1[p] = *(const float4*)&Bb1[(size_t)(k0 + kr) * ldb + n0 + ncq];
      }
    }
  };

  // ---- stage-commit: convert/store register-staged data; fence async copies
  auto stage_commit = [&](int sl) {
    if (!ABF16) {
#pragma unroll
      for (int p = 0; p < 4; ++p) {
        const int qi = tid + 256 * p;
        const int r  = qi >> 3;
        const int cq = (qi & 7) * 4;
        v4bf pk;
        pk[0] = (bf16)aReg[p].x; pk[1] = (bf16)aReg[p].y;
        pk[2] = (bf16)aReg[p].z; pk[3] = (bf16)aReg[p].w;
        *(v4bf*)&As[sl * ASZ + r * LP + cq] = pk;          // ds_store_b64
      }
    }
#pragma unroll
    for (int p = 0; p < 2; ++p) {
      const int qi  = tid + 256 * p;
      const int kr  = qi >> 4;
      const int ncq = (qi & 15) * 4;
      bf16* dst = &Bs0[sl * BSZ + kr];
      dst[(ncq + 0) * LP] = (bf16)bReg0[p].x;
      dst[(ncq + 1) * LP] = (bf16)bReg0[p].y;
      dst[(ncq + 2) * LP] = (bf16)bReg0[p].z;
      dst[(ncq + 3) * LP] = (bf16)bReg0[p].w;
    }
    if (MODE == 0) {
#pragma unroll
      for (int p = 0; p < 2; ++p) {
        const int qi  = tid + 256 * p;
        const int kr  = qi >> 4;
        const int ncq = (qi & 15) * 4;
        bf16* dst = &Bs1[sl * BSZ + kr];
        dst[(ncq + 0) * LP] = (bf16)bReg1[p].x;
        dst[(ncq + 1) * LP] = (bf16)bReg1[p].y;
        dst[(ncq + 2) * LP] = (bf16)bReg1[p].z;
        dst[(ncq + 3) * LP] = (bf16)bReg1[p].w;
      }
    }
    if (ABF16)
      asm volatile("s_wait_asynccnt 0x0" ::: "memory");
  };

  // prologue: fill slot 0
  stage_issue(0, 0);
  stage_commit(0);
  __syncthreads();

  const int h   = lane >> 4;
  const int l16 = lane & 15;

  for (int kt = 0; kt < ntiles; ++kt) {
    const int cur = kt & 1;
    const int nxt = cur ^ 1;
    const bool more = (kt + 1 < ntiles);
    if (more) stage_issue(kt + 1, nxt);

    // A fragment: 16x32 bf16. lane L holds row m=L%16; k = {0..7,16..23} + 8*(L/16)
    const bf16* Ab   = &As[cur * ASZ];
    const bf16* Bb0s = &Bs0[cur * BSZ];
    const bf16* Bb1s = (MODE == 0) ? &Bs1[cur * BSZ] : nullptr;

    v16bf afrag[2];
#pragma unroll
    for (int fm = 0; fm < 2; ++fm) {
      const int r = wm * 32 + fm * 16 + l16;
      union { v16bf v; v8bf p[2]; } u;
      u.p[0] = *(const v8bf*)&Ab[r * LP + 8 * h];
      u.p[1] = *(const v8bf*)&Ab[r * LP + 16 + 8 * h];
      afrag[fm] = u.v;
    }
    // B fragment: 32x16 bf16. lane L holds col n=L%16; k = {0..7,16..23} + 8*(L/16)
    v16bf bfrag0[2], bfrag1[2];
#pragma unroll
    for (int fn = 0; fn < 2; ++fn) {
      const int rn = wn * 32 + fn * 16 + l16;
      union { v16bf v; v8bf p[2]; } u;
      u.p[0] = *(const v8bf*)&Bb0s[rn * LP + 8 * h];
      u.p[1] = *(const v8bf*)&Bb0s[rn * LP + 16 + 8 * h];
      bfrag0[fn] = u.v;
      if (MODE == 0) {
        union { v16bf v; v8bf p[2]; } u1;
        u1.p[0] = *(const v8bf*)&Bb1s[rn * LP + 8 * h];
        u1.p[1] = *(const v8bf*)&Bb1s[rn * LP + 16 + 8 * h];
        bfrag1[fn] = u1.v;
      }
    }
#pragma unroll
    for (int fm = 0; fm < 2; ++fm)
#pragma unroll
      for (int fn = 0; fn < 2; ++fn) {
        accg[fm][fn] = __builtin_amdgcn_wmma_f32_16x16x32_bf16(
            false, afrag[fm], false, bfrag0[fn], (short)0, accg[fm][fn], false, false);
        if (MODE == 0)
          accu[fm][fn] = __builtin_amdgcn_wmma_f32_16x16x32_bf16(
              false, afrag[fm], false, bfrag1[fn], (short)0, accu[fm][fn], false, false);
      }

    if (more) stage_commit(nxt);
    __syncthreads();
  }

  // ---- epilogue; C/D layout: m = vgpr + 8*(lane/16), n = lane%16 ----
#pragma unroll
  for (int fm = 0; fm < 2; ++fm)
#pragma unroll
    for (int fn = 0; fn < 2; ++fn)
#pragma unroll
      for (int i = 0; i < 8; ++i) {
        const int m = m0 + wm * 32 + fm * 16 + h * 8 + i;
        const int n = n0 + wn * 32 + fn * 16 + l16;
        const float g = accg[fm][fn][i];
        if (MODE == 0) {
          float hv = g * fsigmoid(g) * accu[fm][fn][i];     // silu(g) * u
          if (rowscale) hv *= rowscale[(size_t)m * rs_stride + zb * rs_zstep];
          bf16* Cb = (bf16*)Cp + (size_t)zb * Czoff;
          Cb[(size_t)m * ldc + n] = (bf16)hv;
        } else if (MODE == 1) {
          ((float*)Cp)[(size_t)m * ldc + n] = g;
        } else {
          ((float*)Cp)[(size_t)m * ldc + n] =
              addin[(size_t)m * ldc + n] + rowscale[m] * g;
        }
      }
}

// ---------------------------------------------------------------------------
extern "C" void kernel_launch(void* const* d_in, const int* in_sizes, int n_in,
                              void* d_out, int out_size, void* d_ws, size_t ws_size,
                              hipStream_t stream) {
  (void)in_sizes; (void)n_in; (void)out_size; (void)ws_size;

  const float* x   = (const float*)d_in[0];  // [1,2048,2048]
  const float* gw  = (const float*)d_in[1];  // [8,2048]
  const float* Wg  = (const float*)d_in[2];  // [8,2048,1408]
  const float* Wu  = (const float*)d_in[3];  // [8,2048,1408]
  const float* Wd  = (const float*)d_in[4];  // [8,1408,2048]
  const float* Wsg = (const float*)d_in[5];  // [2048,5632]
  const float* Wsu = (const float*)d_in[6];  // [2048,5632]
  const float* Wsd = (const float*)d_in[7];  // [5632,2048]
  const float* sgw = (const float*)d_in[8];  // [2048,1]

  float* out    = (float*)d_out;                       // [T,D] f32
  float* logits = out + (size_t)T_TOK * D_MOD;         // [T,E] f32 (concat)

  // workspace layout
  char*  ws      = (char*)d_ws;
  float* w_dense = (float*)(ws);                                   // T*E   (64 KB)
  float* sgate   = (float*)(ws + 65536);                           // T     (8 KB)
  float* moe     = (float*)(ws + 65536 + 8192);                    // T*D   (16 MB)
  bf16*  big     = (bf16*)(ws + 65536 + 8192 + (size_t)T_TOK * D_MOD * 4);
  // big = max(E*T*F_EXP, T*F_SH) bf16 ~ 46.1 MB; total ws ~ 63 MB

  // 1) router: logits + top-4 dense weights + shared sigmoid gate
  moe_router_kernel<<<T_TOK / 8, 256, 0, stream>>>(x, gw, sgw, w_dense, sgate, logits);

  // 2) expert gate/up, routing weight folded into h' (grid.z = expert)
  {
    dim3 grid(F_EXP / BN, T_TOK / BM, N_EXP);
    wmma_gemm_kernel<0, false><<<grid, 256, 0, stream>>>(
        x, D_MOD, Wg, Wu, F_EXP, big, F_EXP,
        w_dense, N_EXP, /*rs_zstep=*/1, nullptr,
        /*K=*/D_MOD, /*nbatch=*/1, 0L, 0L,
        /*Bzoff=*/(long)D_MOD * F_EXP, /*Czoff=*/(long)T_TOK * F_EXP);
  }
  // 3) down-proj summed over experts: moe = sum_e h'[e] @ Wd[e]
  {
    dim3 grid(D_MOD / BN, T_TOK / BM, 1);
    wmma_gemm_kernel<1, true><<<grid, 256, 0, stream>>>(
        big, F_EXP, Wd, nullptr, D_MOD, moe, D_MOD,
        nullptr, 0, 0, nullptr,
        /*K=*/F_EXP, /*nbatch=*/N_EXP,
        /*Abatch=*/(long)T_TOK * F_EXP, /*Bbatch=*/(long)F_EXP * D_MOD,
        0L, 0L);
  }
  // 4) shared gate/up (reuses big; expert h' is dead after step 3)
  {
    dim3 grid(F_SH / BN, T_TOK / BM, 1);
    wmma_gemm_kernel<0, false><<<grid, 256, 0, stream>>>(
        x, D_MOD, Wsg, Wsu, F_SH, big, F_SH,
        nullptr, 0, 0, nullptr,
        /*K=*/D_MOD, /*nbatch=*/1, 0L, 0L, 0L, 0L);
  }
  // 5) shared down-proj + sigmoid gate + add moe -> d_out
  {
    dim3 grid(D_MOD / BN, T_TOK / BM, 1);
    wmma_gemm_kernel<2, true><<<grid, 256, 0, stream>>>(
        big, F_SH, Wsd, nullptr, D_MOD, out, D_MOD,
        sgate, /*rs_stride=*/1, 0, moe,
        /*K=*/F_SH, /*nbatch=*/1, 0L, 0L, 0L, 0L);
  }
}
